// AtomNet_MP_69252052680908
// MI455X (gfx1250) — compile-verified, matrix operand order checked by reading, not run on hardware
//
#include <hip/hip_runtime.h>
#include <hip/hip_bf16.h>

#define SLOPE 0.2f

typedef __attribute__((ext_vector_type(16))) __bf16 v16bf;
typedef __attribute__((ext_vector_type(8)))  float  v8f;

__device__ __forceinline__ float leakyf(float z) { return z >= 0.f ? z : SLOPE * z; }
__device__ __forceinline__ __bf16 f2bf(float f) { return (__bf16)f; }

// K-index swizzle inside a 16x32 (A) / 32x16 (B) 16-bit WMMA fragment,
// per CDNA5 ISA 7.12.2: VGPR v holds K pair; lanes 0-15 / 16-31 split K blocks.
__device__ __forceinline__ int wmma_k_of(int v, int half, int lo) {
  return ((v < 4) ? (half * 8 + 2 * v) : (16 + half * 8 + 2 * (v - 4))) + lo;
}

__device__ __forceinline__ v8f wmma_bf16(v16bf a, v16bf b, v8f c) {
  // emits v_wmma_f32_16x16x32_bf16
  return __builtin_amdgcn_wmma_f32_16x16x32_bf16(false, a, false, b, (short)0, c,
                                                 false, false);
}

// Load a 16x32 bf16 A-fragment from LDS (row-major, stride STRIDE halves).
// Compiler merges these into ds_load_b128 pairs.
template <int STRIDE>
__device__ __forceinline__ v16bf load_a_lds(const __bf16* s, int k0, int lane) {
  int m = lane & 15, half = lane >> 4;
  const __bf16* row = s + m * STRIDE + k0;
  v16bf a;
#pragma unroll
  for (int v = 0; v < 8; ++v) {
    int k = (v < 4) ? (half * 8 + 2 * v) : (16 + half * 8 + 2 * (v - 4));
    a[2 * v]     = row[k];
    a[2 * v + 1] = row[k + 1];
  }
  return a;
}

// ---------------------------------------------------------------------------
// Pre-swizzle an fp32 weight [R,C] into bf16 WMMA-B fragments, zero padded to
// KT*32 rows x NT*16 cols. dst layout: [(kt*NT+nt)*32 + lane][16 contiguous].
// ---------------------------------------------------------------------------
__global__ void __launch_bounds__(256)
swizzle_b(const float* __restrict__ src, __bf16* __restrict__ dst,
          int R, int C, int NT, int total) {
  int t = blockIdx.x * blockDim.x + threadIdx.x;
  if (t >= total) return;
  int i    = t & 15;
  int lane = (t >> 4) & 31;
  int tile = t >> 9;
  int nt = tile % NT;
  int kt = tile / NT;
  int n = nt * 16 + (lane & 15);
  int k = kt * 32 + wmma_k_of(i >> 1, lane >> 4, i & 1);
  float v = (k < R && n < C) ? src[k * C + n] : 0.f;
  dst[t] = f2bf(v);
}

// ---------------------------------------------------------------------------
// out = leaky(af @ tw1 + b1) @ tw2 + b2      (af: [NA,128], tw: [128,128])
// One block = 16 rows, 8 waves = 8 col-tiles of 16, K = 4 x 32.
// ---------------------------------------------------------------------------
__global__ void __launch_bounds__(256)
feat_transform(const float* __restrict__ af,
               const v16bf* __restrict__ w1sw, const float* __restrict__ b1,
               const v16bf* __restrict__ w2sw, const float* __restrict__ b2,
               float* __restrict__ out) {
  __shared__ __bf16 sA[16][128];
  __shared__ __bf16 sH[16][128];
  int tid = threadIdx.x;
  int lane = tid & 31, wave = tid >> 5;
  int row0 = blockIdx.x * 16;

  for (int e = tid; e < 16 * 128; e += 256) {
    int r = e >> 7, c = e & 127;
    sA[r][c] = f2bf(af[(row0 + r) * 128 + c]);
  }
  __syncthreads();

  int n = lane & 15, half = lane >> 4;
  {
    v8f acc = {};
#pragma unroll
    for (int kk = 0; kk < 4; ++kk) {
      v16bf a = load_a_lds<128>(&sA[0][0], kk * 32, lane);
      v16bf b = w1sw[(kk * 8 + wave) * 32 + lane];
      acc = wmma_bf16(a, b, acc);
    }
    int col = wave * 16 + n;
    float bias = b1[col];
#pragma unroll
    for (int r = 0; r < 8; ++r)
      sH[r + 8 * half][col] = f2bf(leakyf(acc[r] + bias));
  }
  __syncthreads();
  {
    v8f acc = {};
#pragma unroll
    for (int kk = 0; kk < 4; ++kk) {
      v16bf a = load_a_lds<128>(&sH[0][0], kk * 32, lane);
      v16bf b = w2sw[(kk * 8 + wave) * 32 + lane];
      acc = wmma_bf16(a, b, acc);
    }
    int col = wave * 16 + n;
    float bias = b2[col];
#pragma unroll
    for (int r = 0; r < 8; ++r)
      out[(row0 + r + 8 * half) * 128 + col] = acc[r] + bias;
  }
}

// ---------------------------------------------------------------------------
// Batch-masked KNN, LDS-tiled. DB points staged 256 at a time into LDS;
// since db batches are sorted, a whole tile is skipped when the query's
// batch is outside [sb[0], sb[TILE-1]] (~8x work reduction at B=8).
// Top-KK kept in an unrolled register insertion list.
// ---------------------------------------------------------------------------
template <int KK>
__global__ void __launch_bounds__(256)
knn_kernel(const float* __restrict__ q, const float* __restrict__ db,
           const int* __restrict__ qb, const int* __restrict__ dbb,
           int nq, int ndb, int* __restrict__ oidx, float* __restrict__ od,
           int kout, int skip) {
  const int TILE = 256;
  __shared__ float sx[TILE], sy[TILE], sz[TILE];
  __shared__ int   sb[TILE];

  int i = blockIdx.x * blockDim.x + threadIdx.x;
  bool active = i < nq;
  float qx = 0.f, qy = 0.f, qz = 0.f;
  int qbi = -1;
  if (active) {
    qx = q[i * 3]; qy = q[i * 3 + 1]; qz = q[i * 3 + 2];
    qbi = qb[i];
  }
  float bd[KK];
  int bi[KK];
#pragma unroll
  for (int s = 0; s < KK; ++s) { bd[s] = __builtin_inff(); bi[s] = 0; }

  for (int j0 = 0; j0 < ndb; j0 += TILE) {
    __syncthreads();
    {
      int e = threadIdx.x;
      int j = j0 + e;
      if (j < ndb) {
        sx[e] = db[j * 3]; sy[e] = db[j * 3 + 1]; sz[e] = db[j * 3 + 2];
        sb[e] = dbb[j];
      } else {
        sx[e] = 1e30f; sy[e] = 1e30f; sz[e] = 1e30f;
        sb[e] = 0x7fffffff;   // pad: passes range check, fails equality
      }
    }
    __syncthreads();
    // db batches sorted ascending -> tile covers batches [sb[0], sb[TILE-1]]
    if (active && qbi >= sb[0] && qbi <= sb[TILE - 1]) {
      for (int e = 0; e < TILE; ++e) {
        if (sb[e] != qbi) continue;
        float dx = qx - sx[e];
        float dy = qy - sy[e];
        float dz = qz - sz[e];
        float d2 = fmaf(dx, dx, fmaf(dy, dy, dz * dz));
        if (d2 < bd[KK - 1]) {
          bd[KK - 1] = d2; bi[KK - 1] = j0 + e;
#pragma unroll
          for (int s = KK - 1; s > 0; --s) {
            if (bd[s] < bd[s - 1]) {
              float td = bd[s]; bd[s] = bd[s - 1]; bd[s - 1] = td;
              int ti = bi[s]; bi[s] = bi[s - 1]; bi[s - 1] = ti;
            }
          }
        }
      }
    }
  }

  if (active) {
#pragma unroll
    for (int kq = 0; kq < KK; ++kq) {
      if (kq >= skip) {
        oidx[i * kout + (kq - skip)] = bi[kq];
        od[i * kout + (kq - skip)]   = bd[kq];
      }
    }
  }
}

// ---------------------------------------------------------------------------
// One block per node: gather 16 neighbor rows -> feats [16,257] (pad 288),
// h1 = leaky(feats@w1+b1) [16,257] (pad 272/288), h2 = h1@w2+b2 [16,128],
// msg = sum_k h2, out = center + leaky(groupnorm(msg)).
// H=257 => K tiles = 9x32 (rows 257..287 zero), h1 col tiles = 17x16.
// ---------------------------------------------------------------------------
__global__ void __launch_bounds__(256)
mp_layer(const float* __restrict__ center, const float* __restrict__ src,
         const int* __restrict__ idx, const float* __restrict__ dist,
         const v16bf* __restrict__ w1sw, const float* __restrict__ b1,
         const v16bf* __restrict__ w2sw, const float* __restrict__ b2,
         const float* __restrict__ gw, const float* __restrict__ gb,
         float* __restrict__ out) {
  __shared__ __bf16 sF[16][288];
  __shared__ __bf16 sH[16][288];
  __shared__ float  sG[16][128];
  __shared__ int    sIdx[16];
  __shared__ float  sM[128];
  __shared__ float  sStat[4];

  int tid = threadIdx.x;
  int lane = tid & 31, wave = tid >> 5;
  int node = blockIdx.x;

  if (tid < 16) sIdx[tid] = idx[node * 16 + tid];
  for (int e = tid; e < 16 * 32; e += 256) {        // zero feats pad cols 256..287
    int r = e >> 5;
    sF[r][256 + (e & 31)] = f2bf(0.f);
  }
  for (int e = tid; e < 16 * 16; e += 256) {        // zero h1 pad cols 272..287
    int r = e >> 4;
    sH[r][272 + (e & 15)] = f2bf(0.f);
  }
  __syncthreads();

  // feats row k = [center | neighbor_k | dist_k]
  for (int e = tid; e < 16 * 256; e += 256) {
    int r = e >> 8, c = e & 255;
    float v = (c < 128) ? center[node * 128 + c] : src[sIdx[r] * 128 + (c - 128)];
    sF[r][c] = f2bf(v);
  }
  if (tid < 16) sF[tid][256] = f2bf(dist[node * 16 + tid]);
  __syncthreads();

  int ncol = lane & 15, half = lane >> 4;

  // h1: 17 column tiles over 8 waves (wave-uniform loop -> EXEC all-ones)
  for (int nt = wave; nt < 17; nt += 8) {
    v8f acc = {};
#pragma unroll
    for (int kk = 0; kk < 9; ++kk) {
      v16bf a = load_a_lds<288>(&sF[0][0], kk * 32, lane);
      v16bf b = w1sw[(kk * 17 + nt) * 32 + lane];
      acc = wmma_bf16(a, b, acc);
    }
    int col = nt * 16 + ncol;
    float bias = (col < 257) ? b1[col] : 0.f;
#pragma unroll
    for (int r = 0; r < 8; ++r)
      sH[r + 8 * half][col] = f2bf(leakyf(acc[r] + bias));
  }
  __syncthreads();

  // h2: 8 column tiles, one per wave
  {
    v8f acc = {};
#pragma unroll
    for (int kk = 0; kk < 9; ++kk) {
      v16bf a = load_a_lds<288>(&sH[0][0], kk * 32, lane);
      v16bf b = w2sw[(kk * 8 + wave) * 32 + lane];
      acc = wmma_bf16(a, b, acc);
    }
    int col = wave * 16 + ncol;
    float bias = b2[col];
#pragma unroll
    for (int r = 0; r < 8; ++r)
      sG[r + 8 * half][col] = acc[r] + bias;
  }
  __syncthreads();

  // msg = sum over 16 neighbors
  if (tid < 128) {
    float s = 0.f;
#pragma unroll
    for (int r = 0; r < 16; ++r) s += sG[r][tid];
    sM[tid] = s;
  }
  __syncthreads();

  // group norm: 2 groups of 64
  if (tid < 2) {
    float mu = 0.f;
    for (int c = 0; c < 64; ++c) mu += sM[tid * 64 + c];
    mu *= (1.f / 64.f);
    float var = 0.f;
    for (int c = 0; c < 64; ++c) {
      float d = sM[tid * 64 + c] - mu;
      var += d * d;
    }
    var *= (1.f / 64.f);
    sStat[tid * 2]     = mu;
    sStat[tid * 2 + 1] = rsqrtf(var + 1e-5f);
  }
  __syncthreads();
  if (tid < 128) {
    int g = tid >> 6;
    float xn = (sM[tid] - sStat[g * 2]) * sStat[g * 2 + 1];
    float y = leakyf(xn * gw[tid] + gb[tid]);
    out[node * 128 + tid] = center[node * 128 + tid] + y;
  }
}

__global__ void __launch_bounds__(256) fill_ones(float* __restrict__ p, int n) {
  int i = blockIdx.x * blockDim.x + threadIdx.x;
  if (i < n) p[i] = 1.f;
}

// ---------------------------------------------------------------------------
extern "C" void kernel_launch(void* const* d_in, const int* in_sizes, int n_in,
                              void* d_out, int out_size, void* d_ws, size_t ws_size,
                              hipStream_t stream) {
  (void)in_sizes; (void)n_in; (void)out_size; (void)ws_size;
  const int Dd = 128, NA = 4096, NX = 8192, Kn = 16;

  const float* xyz    = (const float*)d_in[0];
  const float* axyz   = (const float*)d_in[1];
  const float* afeat  = (const float*)d_in[2];
  const int*   batch  = (const int*)d_in[3];
  const int*   abatch = (const int*)d_in[4];
  const float* tw1 = (const float*)d_in[5];
  const float* tb1 = (const float*)d_in[6];
  const float* tw2 = (const float*)d_in[7];
  const float* tb2 = (const float*)d_in[8];
  const float* aa_w1 = (const float*)d_in[9];
  const float* aa_b1 = (const float*)d_in[10];
  const float* aa_w2 = (const float*)d_in[11];
  const float* aa_b2 = (const float*)d_in[12];
  const float* aa_gw = (const float*)d_in[13];
  const float* aa_gb = (const float*)d_in[14];
  const float* ae_w1 = (const float*)d_in[15];
  const float* ae_b1 = (const float*)d_in[16];
  const float* ae_w2 = (const float*)d_in[17];
  const float* ae_b2 = (const float*)d_in[18];
  const float* ae_gw = (const float*)d_in[19];
  const float* ae_gb = (const float*)d_in[20];

  char* p = (char*)d_ws;
  auto alloc = [&](size_t bytes) {
    void* r = (void*)p;
    p += (bytes + 255) & ~(size_t)255;
    return r;
  };

  float* outA = (float*)alloc((size_t)NA * Dd * 4);
  float* outB = (float*)alloc((size_t)NA * Dd * 4);
  float* embA = (float*)alloc((size_t)NX * Dd * 4);
  float* embB = (float*)alloc((size_t)NX * Dd * 4);
  int*   idx_aa = (int*)alloc((size_t)NA * Kn * 4);
  float* d_aa   = (float*)alloc((size_t)NA * Kn * 4);
  int*   idx_xa = (int*)alloc((size_t)NX * Kn * 4);
  float* d_xa   = (float*)alloc((size_t)NX * Kn * 4);

  const int T1  = 4 * 8 * 512;    // 128x128 -> 4 K-tiles x 8 N-tiles
  const int W1T = 9 * 17 * 512;   // 257x257 -> 9 x 17
  const int W2T = 9 * 8 * 512;    // 257x128 -> 9 x 8
  v16bf* tw1sw = (v16bf*)alloc((size_t)T1 * 2);
  v16bf* tw2sw = (v16bf*)alloc((size_t)T1 * 2);
  v16bf *aaw1sw[3], *aaw2sw[3], *aew1sw[3], *aew2sw[3];
  for (int i = 0; i < 3; ++i) {
    aaw1sw[i] = (v16bf*)alloc((size_t)W1T * 2);
    aaw2sw[i] = (v16bf*)alloc((size_t)W2T * 2);
  }
  for (int i = 0; i < 3; ++i) {
    aew1sw[i] = (v16bf*)alloc((size_t)W1T * 2);
    aew2sw[i] = (v16bf*)alloc((size_t)W2T * 2);
  }

  // --- pre-swizzle all weights into WMMA-B bf16 fragment layout ---
  swizzle_b<<<(T1 + 255) / 256, 256, 0, stream>>>(tw1, (__bf16*)tw1sw, 128, 128, 8, T1);
  swizzle_b<<<(T1 + 255) / 256, 256, 0, stream>>>(tw2, (__bf16*)tw2sw, 128, 128, 8, T1);
  for (int i = 0; i < 3; ++i) {
    swizzle_b<<<(W1T + 255) / 256, 256, 0, stream>>>(aa_w1 + (size_t)i * 257 * 257,
                                                     (__bf16*)aaw1sw[i], 257, 257, 17, W1T);
    swizzle_b<<<(W2T + 255) / 256, 256, 0, stream>>>(aa_w2 + (size_t)i * 257 * 128,
                                                     (__bf16*)aaw2sw[i], 257, 128, 8, W2T);
    swizzle_b<<<(W1T + 255) / 256, 256, 0, stream>>>(ae_w1 + (size_t)i * 257 * 257,
                                                     (__bf16*)aew1sw[i], 257, 257, 17, W1T);
    swizzle_b<<<(W2T + 255) / 256, 256, 0, stream>>>(ae_w2 + (size_t)i * 257 * 128,
                                                     (__bf16*)aew2sw[i], 257, 128, 8, W2T);
  }

  // --- atom feature transform (WMMA) ---
  feat_transform<<<NA / 16, 256, 0, stream>>>(afeat, tw1sw, tb1, tw2sw, tb2, outA);

  // --- KNN atom->atom (K+1, drop self) ---
  knn_kernel<17><<<(NA + 255) / 256, 256, 0, stream>>>(axyz, axyz, abatch, abatch,
                                                       NA, NA, idx_aa, d_aa, Kn, 1);

  // --- 3 atom->atom message-passing layers ---
  float* cur = outA;
  float* nxt = outB;
  for (int i = 0; i < 3; ++i) {
    mp_layer<<<NA, 256, 0, stream>>>(cur, cur, idx_aa, d_aa,
                                     aaw1sw[i], aa_b1 + i * 257,
                                     aaw2sw[i], aa_b2 + i * 128,
                                     aa_gw + i * 128, aa_gb + i * 128, nxt);
    float* t = cur; cur = nxt; nxt = t;
  }
  float* atom_out = cur;

  // --- KNN point->atom ---
  knn_kernel<16><<<(NX + 255) / 256, 256, 0, stream>>>(xyz, axyz, batch, abatch,
                                                       NX, NA, idx_xa, d_xa, Kn, 0);

  // --- 3 atom->point layers, emb starts at ones, last writes d_out ---
  fill_ones<<<(NX * Dd + 255) / 256, 256, 0, stream>>>(embA, NX * Dd);
  float* ec = embA;
  for (int i = 0; i < 3; ++i) {
    float* dst = (i == 2) ? (float*)d_out : ((ec == embA) ? embB : embA);
    mp_layer<<<NX, 256, 0, stream>>>(ec, atom_out, idx_xa, d_xa,
                                     aew1sw[i], ae_b1 + i * 257,
                                     aew2sw[i], ae_b2 + i * 128,
                                     ae_gw + i * 128, ae_gb + i * 128, dst);
    ec = dst;
  }
}
